// AFGRL_59777354826251
// MI455X (gfx1250) — compile-verified, matrix operand order checked by reading, not run on hardware
//
#include <hip/hip_runtime.h>
#include <hip/hip_bf16.h>

#define NROWS 8192
#define DIM   256
#define NCENT 64
#define NRUNS 5
#define TOPK  8
#define NITER 20
#define CTILE 32   // columns staged per async tile (32 rows x 512B = 16KB)

typedef __attribute__((ext_vector_type(8)))  _Float16 half8;
typedef __attribute__((ext_vector_type(16))) _Float16 half16;
typedef __attribute__((ext_vector_type(8)))  float    float8;

// Load a 16x32 (MxK) f16 WMMA A/B fragment from a row-major [16 x ld] tile.
// ISA 7.12.2: lanes 0-15 -> M=lane, K halves {0..7,16..23}; lanes 16-31 -> {8..15,24..31}.
// For C = X * Y^T both operands load with this same pattern. Works on global or LDS pointers.
__device__ __forceinline__ half16 load_ab_frag(const _Float16* __restrict__ base, int ld, int lane) {
  const _Float16* p = base + (size_t)(lane & 15) * ld + ((lane >> 4) << 3);
  half8 lo = *(const half8*)(p);
  half8 hi = *(const half8*)(p + 16);
  half16 r;
#pragma unroll
  for (int i = 0; i < 8; ++i) { r[i] = lo[i]; r[i + 8] = hi[i]; }
  return r;
}

// ---------------- fp32 -> fp16 conversion ----------------
__global__ void convert_f16_kernel(const float* __restrict__ s, const float* __restrict__ t,
                                   _Float16* __restrict__ s16, _Float16* __restrict__ t16, int n) {
  int i = blockIdx.x * blockDim.x + threadIdx.x;
  if (i < n) { s16[i] = (_Float16)s[i]; t16[i] = (_Float16)t[i]; }
}

// ---------------- k-means: centroid init ----------------
__global__ void kmeans_init_kernel(const float* __restrict__ teacher, const int* __restrict__ init_idx,
                                   float* __restrict__ cent32, _Float16* __restrict__ cent16,
                                   float* __restrict__ cnorm) {
  const int c = blockIdx.x, r = blockIdx.y, d = threadIdx.x;
  const int src = init_idx[r * NCENT + c];
  const float v = teacher[(size_t)src * DIM + d];
  const size_t o = ((size_t)r * NCENT + c) * DIM + d;
  cent32[o] = v; cent16[o] = (_Float16)v;
  __shared__ float red[DIM];
  red[d] = v * v; __syncthreads();
  for (int s2 = DIM / 2; s2 > 0; s2 >>= 1) { if (d < s2) red[d] += red[d + s2]; __syncthreads(); }
  if (d == 0) cnorm[r * NCENT + c] = red[0];
}

// ---------------- k-means: assignment (WMMA: teacher x cent^T, argmin) ----------------
__launch_bounds__(256, 1)
__global__ void kmeans_assign_kernel(const _Float16* __restrict__ T16, const _Float16* __restrict__ cent16,
                                     const float* __restrict__ cnorm, int* __restrict__ labels) {
  __shared__ float tile[8][16 * NCENT];
  const int lane = threadIdx.x & 31;
  const int wave = threadIdx.x >> 5;
  const int run  = blockIdx.y;
  const int row0 = (blockIdx.x * 8 + wave) * 16;
  const _Float16* __restrict__ C16 = cent16 + (size_t)run * NCENT * DIM;
  const float* __restrict__ cn = cnorm + run * NCENT;

  half16 a[8];
#pragma unroll
  for (int kk = 0; kk < 8; ++kk)
    a[kk] = load_ab_frag(T16 + (size_t)row0 * DIM + kk * 32, DIM, lane);

  float8 acc[4];
#pragma unroll
  for (int nt = 0; nt < 4; ++nt)
#pragma unroll
    for (int i = 0; i < 8; ++i) acc[nt][i] = 0.0f;

#pragma unroll
  for (int nt = 0; nt < 4; ++nt) {
    const _Float16* bbase = C16 + (size_t)(nt * 16) * DIM;
#pragma unroll
    for (int kk = 0; kk < 8; ++kk) {
      half16 b = load_ab_frag(bbase + kk * 32, DIM, lane);
      acc[nt] = __builtin_amdgcn_wmma_f32_16x16x32_f16(false, a[kk], false, b,
                                                       (short)0, acc[nt], false, false);
    }
  }
#pragma unroll
  for (int nt = 0; nt < 4; ++nt)
#pragma unroll
    for (int v = 0; v < 8; ++v) {
      const int m  = v + ((lane >> 4) << 3);
      const int nn = (lane & 15) + nt * 16;
      tile[wave][m * NCENT + nn] = acc[nt][v];
    }
  __syncthreads();
  if (lane < 16) {
    float best = 1e30f; int bi = 0;
    for (int nn = 0; nn < NCENT; ++nn) {
      const float d2 = cn[nn] - 2.0f * tile[wave][(lane & 15) * NCENT + nn];
      if (d2 < best) { best = d2; bi = nn; }
    }
    labels[(size_t)run * NROWS + row0 + (lane & 15)] = bi;
  }
}

// ---------------- k-means: centroid update (atomic-free segment sum) ----------------
__global__ void kmeans_update_kernel(const float* __restrict__ teacher, const int* __restrict__ labels,
                                     float* __restrict__ cent32, _Float16* __restrict__ cent16,
                                     float* __restrict__ cnorm) {
  const int c = blockIdx.x, r = blockIdx.y, d = threadIdx.x;
  const int* __restrict__ lab = labels + (size_t)r * NROWS;
  float sum = 0.0f; int count = 0;
  for (int m = 0; m < NROWS; ++m) {
    if (lab[m] == c) { sum += teacher[(size_t)m * DIM + d]; ++count; }  // block-uniform branch
  }
  const size_t o = ((size_t)r * NCENT + c) * DIM + d;
  const float v = (count > 0) ? (sum / (float)count) : cent32[o];
  cent32[o] = v; cent16[o] = (_Float16)v;
  __shared__ float red[DIM];
  red[d] = v * v; __syncthreads();
  for (int s2 = DIM / 2; s2 > 0; s2 >>= 1) { if (d < s2) red[d] += red[d + s2]; __syncthreads(); }
  if (d == 0) cnorm[r * NCENT + c] = red[0];
}

// ---------------- big similarity GEMM + per-row top-8 ----------------
// Teacher column tiles are staged into LDS with double-buffered async copies
// (GLOBAL_LOAD_ASYNC_TO_LDS_B128, ASYNCcnt) and shared by all 8 waves of the block.
__launch_bounds__(256, 1)
__global__ void sim_topk_kernel(const _Float16* __restrict__ S16, const _Float16* __restrict__ T16,
                                float* __restrict__ topv, int* __restrict__ topi) {
  __shared__ __align__(16) _Float16 tbuf[2][CTILE * DIM];  // 2 x 16KB staging
  __shared__ float stile[8][16 * 16];                      // per-wave 16x16 C scratch
  const int lane = threadIdx.x & 31;
  const int wave = threadIdx.x >> 5;
  const int row0 = (blockIdx.x * 8 + wave) * 16;
  const int grow = row0 + (lane & 15);
  const int NT = NROWS / CTILE;

  // per-wave slice of the 16KB tile copy: 4 x b128 per lane
  const unsigned myoff   = (unsigned)(wave * 2048 + lane * 16);
  const unsigned ldsb[2] = { (unsigned)(size_t)(&tbuf[0][0]) + myoff,
                             (unsigned)(size_t)(&tbuf[1][0]) + myoff };
  const unsigned long long gT = (unsigned long long)(size_t)T16;

  half16 a[8];
#pragma unroll
  for (int kk = 0; kk < 8; ++kk)
    a[kk] = load_ab_frag(S16 + (size_t)row0 * DIM + kk * 32, DIM, lane);

  float bestv[TOPK]; int besti[TOPK];
#pragma unroll
  for (int q = 0; q < TOPK; ++q) { bestv[q] = -1e30f; besti[q] = 0; }

  // prologue: async-stage tile 0 into buffer 0
  {
    unsigned long long g = gT + (unsigned long long)myoff;
    unsigned l = ldsb[0];
    asm volatile("global_load_async_to_lds_b128 %0, %1, off"             :: "v"(l), "v"(g) : "memory");
    asm volatile("global_load_async_to_lds_b128 %0, %1, off offset:512"  :: "v"(l), "v"(g) : "memory");
    asm volatile("global_load_async_to_lds_b128 %0, %1, off offset:1024" :: "v"(l), "v"(g) : "memory");
    asm volatile("global_load_async_to_lds_b128 %0, %1, off offset:1536" :: "v"(l), "v"(g) : "memory");
  }

  for (int ct = 0; ct < NT; ++ct) {
    const int buf = ct & 1;
    if (ct + 1 < NT) {
      // issue next tile into the other buffer, then drain only the previous group
      unsigned long long g = gT + (unsigned long long)(ct + 1) * (CTILE * DIM * 2)
                                + (unsigned long long)myoff;
      unsigned l = ldsb[buf ^ 1];
      asm volatile("global_load_async_to_lds_b128 %0, %1, off"             :: "v"(l), "v"(g) : "memory");
      asm volatile("global_load_async_to_lds_b128 %0, %1, off offset:512"  :: "v"(l), "v"(g) : "memory");
      asm volatile("global_load_async_to_lds_b128 %0, %1, off offset:1024" :: "v"(l), "v"(g) : "memory");
      asm volatile("global_load_async_to_lds_b128 %0, %1, off offset:1536" :: "v"(l), "v"(g) : "memory");
      asm volatile("s_wait_asynccnt 0x4" ::: "memory");
    } else {
      asm volatile("s_wait_asynccnt 0x0" ::: "memory");
    }
    __syncthreads();  // all waves' slices of tile `ct` have landed in LDS

    const _Float16* tb = &tbuf[buf][0];
#pragma unroll
    for (int nt = 0; nt < 2; ++nt) {
      float8 acc;
#pragma unroll
      for (int i = 0; i < 8; ++i) acc[i] = 0.0f;
#pragma unroll
      for (int kk = 0; kk < 8; ++kk) {
        half16 b = load_ab_frag(tb + (size_t)(nt * 16) * DIM + kk * 32, DIM, lane);  // ds_load_b128
        acc = __builtin_amdgcn_wmma_f32_16x16x32_f16(false, a[kk], false, b,
                                                     (short)0, acc, false, false);
      }
#pragma unroll
      for (int v = 0; v < 8; ++v)
        stile[wave][(v + ((lane >> 4) << 3)) * 16 + (lane & 15)] = acc[v];
      __syncthreads();
      if (lane < 16) {
        const int cbase = ct * CTILE + nt * 16;
#pragma unroll
        for (int nn = 0; nn < 16; ++nn) {
          float v = stile[wave][(lane & 15) * 16 + nn];
          const int col = cbase + nn;
          if (col == grow) v += 10.0f;               // diagonal boost
          if (v > bestv[TOPK - 1]) {                 // sorted insert, static indices only
            bestv[TOPK - 1] = v; besti[TOPK - 1] = col;
#pragma unroll
            for (int q = TOPK - 1; q >= 1; --q) {
              if (bestv[q] > bestv[q - 1]) {
                float tv = bestv[q]; bestv[q] = bestv[q - 1]; bestv[q - 1] = tv;
                int  ti = besti[q]; besti[q] = besti[q - 1]; besti[q - 1] = ti;
              }
            }
          }
        }
      }
      __syncthreads();  // also guards buffer reuse before the next async issue
    }
  }
  if (lane < 16) {
#pragma unroll
    for (int q = 0; q < TOPK; ++q) {
      topv[(size_t)grow * TOPK + q] = bestv[q];
      topi[(size_t)grow * TOPK + q] = besti[q];
    }
  }
}

// ---------------- adjacency bucketing ----------------
__global__ void zero2_kernel(int* __restrict__ a, int* __restrict__ b, int n) {
  int i = blockIdx.x * blockDim.x + threadIdx.x;
  if (i < n) { a[i] = 0; b[i] = 0; }
}

__global__ void edge_count_kernel(const int* __restrict__ src, int* __restrict__ cnt, int E) {
  int e = blockIdx.x * blockDim.x + threadIdx.x;
  if (e < E) atomicAdd(&cnt[src[e]], 1);
}

__global__ void scan_kernel(const int* __restrict__ cnt, int* __restrict__ offs) {
  __shared__ int part[256];
  const int t = threadIdx.x;
  const int base = t * (NROWS / 256);
  int s = 0;
  for (int i = 0; i < NROWS / 256; ++i) s += cnt[base + i];
  part[t] = s; __syncthreads();
  for (int d = 1; d < 256; d <<= 1) {
    int v = (t >= d) ? part[t - d] : 0;
    __syncthreads();
    part[t] += v;
    __syncthreads();
  }
  int run = (t == 0) ? 0 : part[t - 1];
  for (int i = 0; i < NROWS / 256; ++i) { offs[base + i] = run; run += cnt[base + i]; }
}

__global__ void edge_scatter_kernel(const int* __restrict__ src, const int* __restrict__ dst,
                                    const int* __restrict__ offs, int* __restrict__ cursor,
                                    int* __restrict__ adjcol, int E) {
  int e = blockIdx.x * blockDim.x + threadIdx.x;
  if (e < E) {
    const int s = src[e];
    const int p = atomicAdd(&cursor[s], 1);
    adjcol[offs[s] + p] = dst[e];
  }
}

// ---------------- final mask + output assembly ----------------
__global__ void finalize_kernel(const int* __restrict__ topi, const float* __restrict__ topv,
                                const int* __restrict__ offs, const int* __restrict__ cnt,
                                const int* __restrict__ adjcol, const int* __restrict__ labels,
                                float* __restrict__ out) {
  const int t = blockIdx.x * blockDim.x + threadIdx.x;
  if (t >= NROWS * TOPK) return;
  const int i = t >> 3;
  const int j = topi[t];
  bool in_adj = false;
  const int b0 = offs[i], c = cnt[i];
  for (int q = 0; q < c; ++q) in_adj = in_adj || (adjcol[b0 + q] == j);
  bool close = false;
#pragma unroll
  for (int r = 0; r < NRUNS; ++r)
    close = close || (labels[(size_t)r * NROWS + i] == labels[(size_t)r * NROWS + j]);
  const int NK = NROWS * TOPK;
  out[t]          = (float)j;                        // I_knn
  out[NK + t]     = (in_adj || close) ? 1.0f : 0.0f; // pos_mask
  out[2 * NK + t] = topv[t];                         // D_knn
}

extern "C" void kernel_launch(void* const* d_in, const int* in_sizes, int n_in,
                              void* d_out, int out_size, void* d_ws, size_t ws_size,
                              hipStream_t stream) {
  const float* student = (const float*)d_in[0];
  const float* teacher = (const float*)d_in[1];
  const int*   edge    = (const int*)d_in[2];
  const int*   initix  = (const int*)d_in[3];
  const int E = in_sizes[2] / 2;
  const int* esrc = edge;
  const int* edst = edge + E;

  char* ws = (char*)d_ws;
  auto carve = [&](size_t bytes) -> char* {
    char* p = ws; ws += (bytes + 255) & ~(size_t)255; return p;
  };
  _Float16* S16    = (_Float16*)carve((size_t)NROWS * DIM * 2);
  _Float16* T16    = (_Float16*)carve((size_t)NROWS * DIM * 2);
  float*    cent32 = (float*)   carve((size_t)NRUNS * NCENT * DIM * 4);
  _Float16* cent16 = (_Float16*)carve((size_t)NRUNS * NCENT * DIM * 2);
  float*    cnorm  = (float*)   carve((size_t)NRUNS * NCENT * 4);
  int*      labels = (int*)     carve((size_t)NRUNS * NROWS * 4);
  float*    topv   = (float*)   carve((size_t)NROWS * TOPK * 4);
  int*      topi   = (int*)     carve((size_t)NROWS * TOPK * 4);
  int*      cnt    = (int*)     carve((size_t)NROWS * 4);
  int*      offs   = (int*)     carve((size_t)NROWS * 4);
  int*      cursor = (int*)     carve((size_t)NROWS * 4);
  int*      adjcol = (int*)     carve((size_t)E * 4);

  // 1) convert to f16
  {
    const int n = NROWS * DIM;
    convert_f16_kernel<<<(n + 255) / 256, 256, 0, stream>>>(student, teacher, S16, T16, n);
  }
  // 2) k-means: init, 20 x (assign, update), final assign
  kmeans_init_kernel<<<dim3(NCENT, NRUNS), DIM, 0, stream>>>(teacher, initix, cent32, cent16, cnorm);
  for (int it = 0; it < NITER; ++it) {
    kmeans_assign_kernel<<<dim3(NROWS / 128, NRUNS), 256, 0, stream>>>(T16, cent16, cnorm, labels);
    kmeans_update_kernel<<<dim3(NCENT, NRUNS), DIM, 0, stream>>>(teacher, labels, cent32, cent16, cnorm);
  }
  kmeans_assign_kernel<<<dim3(NROWS / 128, NRUNS), 256, 0, stream>>>(T16, cent16, cnorm, labels);

  // 3) adjacency bucketing
  zero2_kernel<<<(NROWS + 255) / 256, 256, 0, stream>>>(cnt, cursor, NROWS);
  edge_count_kernel<<<(E + 255) / 256, 256, 0, stream>>>(esrc, cnt, E);
  scan_kernel<<<1, 256, 0, stream>>>(cnt, offs);
  edge_scatter_kernel<<<(E + 255) / 256, 256, 0, stream>>>(esrc, edst, offs, cursor, adjcol, E);

  // 4) similarity GEMM + top-8 (async-LDS double-buffered)
  sim_topk_kernel<<<NROWS / 128, 256, 0, stream>>>(S16, T16, topv, topi);

  // 5) masks + outputs
  finalize_kernel<<<(NROWS * TOPK + 255) / 256, 256, 0, stream>>>(topi, topv, offs, cnt, adjcol, labels, (float*)d_out);
}
